// CapsuleLayer_6717328851113
// MI455X (gfx1250) — matrix operand order repaired and checked by reading, not compiled, not run on hardware
//
#include <hip/hip_runtime.h>
#include <hip/hip_bf16.h>

// ---------------------------------------------------------------------------
// CapsuleLayer on MI455X (gfx1250).
//   u = x @ W  : (65536 x 256) x (256 x 512), bf16 WMMA, f32 accumulate.
//   3 dynamic-routing iterations; iter 0 fused into GEMM epilogue (c = 1/32).
//   u stored bf16 (64 MB) -> resident in 192 MB L2 for the routing passes.
//   W tiles staged to LDS via global_load_async_to_lds_b128 (ASYNCcnt path);
//   s reductions pre-reduced in LDS (ds_add_f32) to kill atomic contention.
// ---------------------------------------------------------------------------

typedef __attribute__((ext_vector_type(16))) __bf16 v16bf;
typedef __attribute__((ext_vector_type(8)))  float  v8f;

#define B_    32
#define NI_   2048
#define D_    256
#define CAPS  32
#define DIM   16
#define COLS  512                 // CAPS*DIM
#define MROWS (B_ * NI_)          // 65536
#define KBLK  (D_ / 32)           // 8 k-tiles of 32
#define MBLK  (MROWS / 16)        // 4096 16-row tiles
#define CBLK  (COLS / 16)         // 32 16-col tiles
#define EPSQ  1e-7f

// ------------------------------ utility ------------------------------------

__global__ void caps_zero_f32(float* __restrict__ p, int n) {
  int i = blockIdx.x * blockDim.x + threadIdx.x;
  if (i < n) p[i] = 0.0f;
}

__device__ inline void caps_unpack2(unsigned int w, float& lo, float& hi) {
  lo = __uint_as_float(w << 16);           // bf16 -> f32 = bits << 16
  hi = __uint_as_float(w & 0xffff0000u);
}

// ---------------- pack A into WMMA A-operand register layout ----------------
// ISA 7.12.2, 16-bit A 16x32: lane<16 : K = {0..7, 16..23}
//                             lane>=16: K = {8..15, 24..31}
// Slot t = (mblk*KBLK + kblk)*32 + lane holds 16 bf16 (32B, contiguous).
__global__ void caps_pack_a(const float* __restrict__ x, __bf16* __restrict__ dst) {
  int t    = blockIdx.x * blockDim.x + threadIdx.x;   // MBLK*KBLK*32 threads
  int mblk = t >> 8;                                  // KBLK*32 = 256 slots/mblk
  int slot = t & 255;
  int kblk = slot >> 5;
  int lane = slot & 31;
  int row   = (mblk << 4) + (lane & 15);
  int kbase = (kblk << 5) + ((lane >> 4) << 3);       // +0 or +8
  const float* src = x + (size_t)row * D_ + kbase;
  __bf16* o = dst + ((size_t)t << 4);
#pragma unroll
  for (int e = 0; e < 8; ++e) o[e]     = (__bf16)src[e];        // K = kbase+e
#pragma unroll
  for (int e = 0; e < 8; ++e) o[8 + e] = (__bf16)src[16 + e];   // K = kbase+16+e
}

// ---------------- pack W into WMMA B-operand register layout ----------------
// 32x16 bf16 B: lane = column (N), lanes 0-15 hold K={0..7,16..23}, 16-31 hold K={8..15,24..31}.
__global__ void caps_pack_w(const float* __restrict__ w, __bf16* __restrict__ dst) {
  int t = blockIdx.x * blockDim.x + threadIdx.x;      // CBLK*KBLK*32 = 8192
  if (t >= CBLK * KBLK * 32) return;
  int cblk = t >> 8;
  int slot = t & 255;
  int kblk = slot >> 5;
  int lane = slot & 31;
  int col   = (cblk << 4) + (lane & 15);
  int kbase = (kblk << 5) + ((lane >> 4) << 3);       // +0 or +8
  __bf16* o = dst + ((size_t)t << 4);
#pragma unroll
  for (int e = 0; e < 16; ++e)
    o[e] = (__bf16)w[(size_t)(kbase + (e & 7) + ((e >> 3) << 4)) * COLS + col];
}

// ------------------------------ WMMA GEMM -----------------------------------
// Workgroup: 256 threads = 8 waves, tile 64(M) x 128(N); wave tile 16 x 64.
// W tiles for this block's 128 columns staged into LDS in two 32KB K-phases
// via async-to-LDS loads; B operand then fed from ds_load (DScnt), while A
// streams through VMEM (LOADcnt) -> decoupled wait chains.
// Epilogue: store u (bf16) + fused routing iter-0 (c = 1/32), LDS-pre-reduced.
__global__ __launch_bounds__(256)
void caps_gemm_wmma(const __bf16* __restrict__ Ap,
                    const __bf16* __restrict__ Wp,
                    __bf16* __restrict__ u,
                    float*  __restrict__ s0) {
  __shared__ __align__(32) __bf16 lds_w[16384];   // 32KB: 8 colblk x 4 kblk x 1KB
  __shared__ float sacc[128];                     // per-block column partials

  const int lane  = threadIdx.x & 31;
  const int wv    = threadIdx.x >> 5;          // 0..7
  const int wm    = wv >> 1;                   // 0..3 along M
  const int wn    = wv & 1;                    // 0..1 along N
  const int mblk  = blockIdx.x * 4 + wm;       // 16-row tile id
  const int cblk0 = blockIdx.y * 8 + wn * 4;   // first of 4 16-col tiles

  for (int i = threadIdx.x; i < 128; i += 256) sacc[i] = 0.0f;

  v8f acc[4] = {};
  const v16bf*   a_base   = (const v16bf*)Ap + (size_t)mblk * KBLK * 32 + lane;
  const char*    wbytes   = (const char*)Wp;
  const unsigned lds_base = (unsigned)(unsigned long long)(&lds_w[0]);

  for (int p = 0; p < 2; ++p) {                // two K-phases of 4 kblks
    __syncthreads();                           // previous-phase readers done
    // Stage 32KB of W: 2048 x 16B chunks, 8 per thread, async to LDS.
#pragma unroll
    for (int it = 0; it < 8; ++it) {
      const int li = it * 256 + threadIdx.x;   // 16B-chunk index in phase buf
      const int tt = li >> 6;                  // 1KB tile id: cb*4 + kk
      const int c  = li & 63;                  // 16B chunk within tile
      const int cb = tt >> 2, kk = tt & 3;
      const char* src = wbytes +
          (((size_t)(blockIdx.y * 8 + cb) * KBLK + (size_t)(p * 4 + kk)) << 10) +
          ((size_t)c << 4);
      unsigned dst = lds_base + (unsigned)(li << 4);
      asm volatile("global_load_async_to_lds_b128 %0, %1, off"
                   :: "v"(dst), "v"(src) : "memory");
    }
    asm volatile("s_wait_asynccnt 0x0" ::: "memory");
    __syncthreads();                           // all waves' staging visible

#pragma unroll
    for (int kk = 0; kk < 4; ++kk) {
      const int kb = p * 4 + kk;
      v16bf a = a_base[(size_t)kb * 32];
      if (kb + 1 < KBLK)
        __builtin_prefetch((const void*)(a_base + (size_t)(kb + 1) * 32), 0, 1);
#pragma unroll
      for (int t = 0; t < 4; ++t) {
        v16bf b = ((const v16bf*)lds_w)[(size_t)(((wn * 4 + t) * 4 + kk) * 32 + lane)];
        acc[t] = __builtin_amdgcn_wmma_f32_16x16x32_bf16(
            /*neg_a=*/false, a, /*neg_b=*/false, b,
            /*c_mod=*/(short)0, acc[t], /*reuse_a=*/false, /*reuse_b=*/false);
      }
    }
  }

  // C layout: VGPR r, lanes 0-15: row=r, col=lane; lanes 16-31: row=r+8.
  const int row0 = (mblk << 4) + ((lane >> 4) << 3);
#pragma unroll
  for (int t = 0; t < 4; ++t) {
    const int col = ((cblk0 + t) << 4) + (lane & 15);
    float csum = 0.0f;
#pragma unroll
    for (int r = 0; r < 8; ++r) {
      float v = acc[t][r];
      csum += v;
      u[(size_t)(row0 + r) * COLS + col] = (__bf16)v;
    }
    csum += __shfl_xor(csum, 16, 32);          // fold the two 8-row halves
    if (lane < 16)                             // ds_add_f32 pre-reduction
      atomicAdd(&sacc[(wn << 6) + (t << 4) + lane], csum);
  }
  __syncthreads();
  const int bb = blockIdx.x >> 5;              // batch (uniform per block)
  for (int i = threadIdx.x; i < 128; i += 256)
    atomicAdd(&s0[(size_t)bb * COLS + ((size_t)blockIdx.y << 7) + i],
              sacc[i] * (1.0f / CAPS));
}

// ------------------------------- squash -------------------------------------
__global__ void caps_squash(const float* __restrict__ s, float* __restrict__ v) {
  int t = blockIdx.x * blockDim.x + threadIdx.x;      // B_*CAPS = 1024
  if (t >= B_ * CAPS) return;
  const float* sp = s + (size_t)t * DIM;
  float tmp[DIM];
  float s2 = 0.0f;
#pragma unroll
  for (int d = 0; d < DIM; ++d) { tmp[d] = sp[d]; s2 += tmp[d] * tmp[d]; }
  float scale = s2 / ((1.0f + s2) * sqrtf(s2 + EPSQ));
  float* vp = v + (size_t)t * DIM;
#pragma unroll
  for (int d = 0; d < DIM; ++d) vp[d] = scale * tmp[d];
}

// ------------------------------- routing ------------------------------------
// One wave per i-row strip; lane = capsule. Fuses b-update (with previous v),
// wave-wide softmax over the 32 capsules, and c*u accumulation; the s partials
// are pre-reduced across the block's 8 waves in LDS before global atomics.
__global__ __launch_bounds__(256)
void caps_route(const __bf16* __restrict__ u,
                const float*  __restrict__ v_prev,   // (B,CAPS,DIM)
                float* __restrict__ b_buf,           // (B,NI,CAPS)
                float* __restrict__ s_out,           // (B,CAPS,DIM)
                int b_valid, int write_b) {
  __shared__ float rsacc[COLS];                      // 512 f32
  const int lane  = threadIdx.x & 31;                // capsule
  const int wv    = threadIdx.x >> 5;
  const int bb    = blockIdx.x;
  const int strip = blockIdx.y * 8 + wv;             // 0..127
  const int ipw   = NI_ / 128;                       // 16 rows per wave
  const int i0    = strip * ipw;

  for (int i = threadIdx.x; i < COLS; i += 256) rsacc[i] = 0.0f;
  __syncthreads();

  float vr[DIM];
#pragma unroll
  for (int d = 0; d < DIM; ++d)
    vr[d] = v_prev[((size_t)bb * CAPS + lane) * DIM + d];

  float accs[DIM];
#pragma unroll
  for (int d = 0; d < DIM; ++d) accs[d] = 0.0f;

#pragma unroll 2
  for (int ii = 0; ii < ipw; ++ii) {
    const int i = i0 + ii;
    const size_t urow = ((size_t)(bb * NI_ + i) * CAPS + lane) * DIM;
    const uint4* up = (const uint4*)(u + urow);      // 32B/lane, coalesced 1KB
    uint4 q0 = up[0], q1 = up[1];
    float uf[DIM];
    caps_unpack2(q0.x, uf[0],  uf[1]);  caps_unpack2(q0.y, uf[2],  uf[3]);
    caps_unpack2(q0.z, uf[4],  uf[5]);  caps_unpack2(q0.w, uf[6],  uf[7]);
    caps_unpack2(q1.x, uf[8],  uf[9]);  caps_unpack2(q1.y, uf[10], uf[11]);
    caps_unpack2(q1.z, uf[12], uf[13]); caps_unpack2(q1.w, uf[14], uf[15]);

    float dot = 0.0f;
#pragma unroll
    for (int d = 0; d < DIM; ++d) dot += uf[d] * vr[d];

    const size_t bidx = (size_t)(bb * NI_ + i) * CAPS + lane;
    float logit = dot + (b_valid ? b_buf[bidx] : 0.0f);
    if (write_b) b_buf[bidx] = logit;

    // softmax over 32 capsules (wave32 lanes)
    float m = logit;
#pragma unroll
    for (int o = 16; o > 0; o >>= 1) m = fmaxf(m, __shfl_xor(m, o, 32));
    float e = __expf(logit - m);
    float den = e;
#pragma unroll
    for (int o = 16; o > 0; o >>= 1) den += __shfl_xor(den, o, 32);
    float c = e / den;
#pragma unroll
    for (int d = 0; d < DIM; ++d) accs[d] += c * uf[d];
  }

  // pre-reduce across the block's 8 waves in LDS, then one atomic per slot
#pragma unroll
  for (int d = 0; d < DIM; ++d)
    atomicAdd(&rsacc[lane * DIM + d], accs[d]);
  __syncthreads();
  for (int i = threadIdx.x; i < COLS; i += 256)
    atomicAdd(&s_out[(size_t)bb * COLS + i], rsacc[i]);
}

// ------------------------------ launcher ------------------------------------

extern "C" void kernel_launch(void* const* d_in, const int* in_sizes, int n_in,
                              void* d_out, int out_size, void* d_ws, size_t ws_size,
                              hipStream_t stream) {
  (void)in_sizes; (void)n_in; (void)out_size; (void)ws_size;
  const float* x = (const float*)d_in[0];   // (32, 2048, 256) f32
  const float* W = (const float*)d_in[1];   // (256, 512) f32
  float* out = (float*)d_out;               // (32, 32, 16) f32

  char* ws = (char*)d_ws;
  size_t off = 0;
  auto wsalloc = [&](size_t bytes) -> char* {
    char* p = ws + off;
    off += (bytes + 255) & ~(size_t)255;
    return p;
  };
  __bf16* Ap = (__bf16*)wsalloc((size_t)MROWS * D_ * 2);      // 32 MB
  __bf16* Wp = (__bf16*)wsalloc((size_t)D_ * COLS * 2);       // 256 KB
  __bf16* U  = (__bf16*)wsalloc((size_t)MROWS * COLS * 2);    // 64 MB
  float*  Bb = (float*)wsalloc((size_t)B_ * NI_ * CAPS * 4);  // 8 MB
  float*  S  = (float*)wsalloc((size_t)3 * B_ * COLS * 4);    // s0|s1|s2
  float*  S0 = S;
  float*  S1 = S + (size_t)B_ * COLS;
  float*  S2 = S + (size_t)2 * B_ * COLS;
  float*  V0 = (float*)wsalloc((size_t)B_ * COLS * 4);
  float*  V1 = (float*)wsalloc((size_t)B_ * COLS * 4);

  // 1) zero the three s accumulators (ws is poisoned by the harness)
  const int nz = 3 * B_ * COLS;
  caps_zero_f32<<<(nz + 255) / 256, 256, 0, stream>>>(S, nz);

  // 2) bf16 downcast + WMMA operand swizzle
  caps_pack_a<<<(MBLK * KBLK * 32) / 256, 256, 0, stream>>>(x, Ap);
  caps_pack_w<<<(CBLK * KBLK * 32 + 255) / 256, 256, 0, stream>>>(W, Wp);

  // 3) projection GEMM + fused routing iteration 0 (uniform c = 1/32)
  dim3 gg(MBLK / 4, CBLK / 8);   // 1024 x 4 workgroups
  caps_gemm_wmma<<<gg, 256, 0, stream>>>(Ap, Wp, U, S0);
  caps_squash<<<(B_ * CAPS + 255) / 256, 256, 0, stream>>>(S0, V0);

  // 4) routing iterations 1 and 2 (each reads u exactly once, L2-resident)
  dim3 rg(B_, 16);
  caps_route<<<rg, 256, 0, stream>>>(U, V0, Bb, S1, /*b_valid=*/0, /*write_b=*/1);
  caps_squash<<<(B_ * CAPS + 255) / 256, 256, 0, stream>>>(S1, V1);
  caps_route<<<rg, 256, 0, stream>>>(U, V1, Bb, S2, /*b_valid=*/1, /*write_b=*/0);
  caps_squash<<<(B_ * CAPS + 255) / 256, 256, 0, stream>>>(S2, out);
}
